// RelPositionMultiheadAttention_13348758356019
// MI455X (gfx1250) — compile-verified, matrix operand order checked by reading, not run on hardware
//
#include <hip/hip_runtime.h>
#include <hip/hip_bf16.h>

// ---------------------------------------------------------------------------
// RelPositionMultiheadAttention for MI455X (gfx1250, wave32, WMMA f16->f32).
// L=2048, N=4, E=256, H=4, D=64.
//
//   1. proj_qkv : qkv = x @ Win^T + b ; writes Qu,Qv,K (f16, [n][h][L][D]) and
//                 V transposed (f16, [n][h][D][L]) for contiguous B-fragments.
//   2. proj_pos : p = pos_emb @ Wp^T ; f16 [h][2L-1(pad 4096)][D].
//   3. attn     : flash-style streaming softmax(QuK^T + relshift(QvP^T)) V.
//   4. out_proj : out = O @ Wo^T + bo (fp32 output).
//
// All GEMM waves compute 16x32 tiles (two interleaved WMMA accumulator chains
// sharing one A fragment) to hide WMMA->WMMA RAW latency and halve the
// f32->f16 conversion VALU work per output element.
// ---------------------------------------------------------------------------

typedef __attribute__((ext_vector_type(16))) _Float16 v16h;
typedef __attribute__((ext_vector_type(8)))  _Float16 h8v;
typedef __attribute__((ext_vector_type(8)))  float    v8f;
typedef __attribute__((ext_vector_type(8)))  float    f8v;

constexpr int Lseq = 2048;
constexpr int NBAT = 4;
constexpr int EMB  = 256;
constexpr int NH   = 4;
constexpr int HD   = 64;
constexpr int PROWS = 2 * Lseq - 1;   // 4095
constexpr int PPAD  = 4096;           // padded row count for P

__device__ __forceinline__ v8f wmma_f16(v16h a, v16h b, v8f c) {
  return __builtin_amdgcn_wmma_f32_16x16x32_f16(false, a, false, b, (short)0, c,
                                                false, false);
}

// A/B fragment for V_WMMA_F32_16X16X32_F16 from row-major f16 with contiguous K.
// Lane m<16 holds row m, K = {0..7, 16..23}; lane m+16 holds row m, K = {8..15, 24..31}.
__device__ __forceinline__ v16h frag_f16(const _Float16* base, int ld, int r0,
                                         int k0, int lane) {
  const int m = lane & 15, half = lane >> 4;
  const _Float16* p = base + (size_t)(r0 + m) * ld + k0 + half * 8;
  h8v lo = *(const h8v*)(p);
  h8v hi = *(const h8v*)(p + 16);
  v16h f;
#pragma unroll
  for (int t = 0; t < 8; ++t) { f[t] = lo[t]; f[t + 8] = hi[t]; }
  return f;
}

__device__ __forceinline__ v16h frag_f16_clamp(const _Float16* base, int ld,
                                               int r0, int k0, int lane, int rmax) {
  const int m = lane & 15, half = lane >> 4;
  int row = r0 + m;
  row = row < 0 ? 0 : (row > rmax ? rmax : row);
  const _Float16* p = base + (size_t)row * ld + k0 + half * 8;
  h8v lo = *(const h8v*)(p);
  h8v hi = *(const h8v*)(p + 16);
  v16h f;
#pragma unroll
  for (int t = 0; t < 8; ++t) { f[t] = lo[t]; f[t + 8] = hi[t]; }
  return f;
}

__device__ __forceinline__ v16h frag_f32cvt(const float* base, int ld, int r0,
                                            int k0, int lane) {
  const int m = lane & 15, half = lane >> 4;
  const float* p = base + (size_t)(r0 + m) * ld + k0 + half * 8;
  f8v lo = *(const f8v*)(p);
  f8v hi = *(const f8v*)(p + 16);
  v16h f;
#pragma unroll
  for (int t = 0; t < 8; ++t) { f[t] = (_Float16)lo[t]; f[t + 8] = (_Float16)hi[t]; }
  return f;
}

__device__ __forceinline__ v16h frag_f32cvt_clamp(const float* base, int ld,
                                                  int r0, int k0, int lane,
                                                  int rmax) {
  const int m = lane & 15, half = lane >> 4;
  int row = r0 + m;
  row = row < 0 ? 0 : (row > rmax ? rmax : row);
  const float* p = base + (size_t)row * ld + k0 + half * 8;
  f8v lo = *(const f8v*)(p);
  f8v hi = *(const f8v*)(p + 16);
  v16h f;
#pragma unroll
  for (int t = 0; t < 8; ++t) { f[t] = (_Float16)lo[t]; f[t + 8] = (_Float16)hi[t]; }
  return f;
}

// ---------------------------------------------------------------------------
// Kernel 1: QKV projection. 16x32 tiles: (8192/16)=512 x (768/32)=24 waves.
// ---------------------------------------------------------------------------
__device__ __forceinline__ void qkv_store(float v, int R, int C,
                                          const float* bIn, const float* bu,
                                          const float* bv, _Float16* Qu,
                                          _Float16* Qv, _Float16* Kb,
                                          _Float16* Vt) {
  v += bIn[C];
  const int i = R >> 2, nb = R & 3;                 // flat row = i*NBAT + nb
  const int sec = C >> 8, ch = C & 255;
  const int h = ch >> 6, d = ch & 63;
  const size_t idx = (((size_t)(nb * NH + h)) * Lseq + i) * HD + d;
  if (sec == 0) {
    v *= 0.125f;                                    // 1/sqrt(D)
    Qu[idx] = (_Float16)(v + bu[ch]);
    Qv[idx] = (_Float16)(v + bv[ch]);
  } else if (sec == 1) {
    Kb[idx] = (_Float16)v;
  } else {
    Vt[(((size_t)(nb * NH + h)) * HD + d) * Lseq + i] = (_Float16)v;
  }
}

__global__ void __launch_bounds__(128) proj_qkv(
    const float* __restrict__ x, const float* __restrict__ wIn,
    const float* __restrict__ bIn, const float* __restrict__ bu,
    const float* __restrict__ bv, _Float16* __restrict__ Qu,
    _Float16* __restrict__ Qv, _Float16* __restrict__ Kb,
    _Float16* __restrict__ Vt) {
  const int w = threadIdx.x >> 5, lane = threadIdx.x & 31;
  const int tile = blockIdx.x * 4 + w;
  const int mt = tile / 24, nt = tile % 24;
  const int m0 = mt * 16, n0 = nt * 32;

  v8f acc0 = {}, acc1 = {};
#pragma unroll
  for (int k0 = 0; k0 < EMB; k0 += 32) {
    v16h a  = frag_f32cvt(x, EMB, m0, k0, lane);
    v16h b0 = frag_f32cvt(wIn, EMB, n0, k0, lane);       // B(k,n) = Win[n][k]
    v16h b1 = frag_f32cvt(wIn, EMB, n0 + 16, k0, lane);
    acc0 = wmma_f16(a, b0, acc0);
    acc1 = wmma_f16(a, b1, acc1);
  }

  const int nl = lane & 15, M0 = (lane >> 4) * 8;
#pragma unroll
  for (int r = 0; r < 8; ++r) {
    const int R = m0 + M0 + r;
    qkv_store(acc0[r], R, n0 + nl,      bIn, bu, bv, Qu, Qv, Kb, Vt);
    qkv_store(acc1[r], R, n0 + 16 + nl, bIn, bu, bv, Qu, Qv, Kb, Vt);
  }
}

// ---------------------------------------------------------------------------
// Kernel 2: positional projection. 16x32 tiles: (4096/16)=256 x (256/32)=8.
// ---------------------------------------------------------------------------
__global__ void __launch_bounds__(128) proj_pos(const float* __restrict__ pe,
                                                const float* __restrict__ wP,
                                                _Float16* __restrict__ Pb) {
  const int w = threadIdx.x >> 5, lane = threadIdx.x & 31;
  const int tile = blockIdx.x * 4 + w;
  const int mt = tile / 8, nt = tile % 8;
  const int m0 = mt * 16, n0 = nt * 32;

  v8f acc0 = {}, acc1 = {};
#pragma unroll
  for (int k0 = 0; k0 < EMB; k0 += 32) {
    v16h a  = frag_f32cvt_clamp(pe, EMB, m0, k0, lane, PROWS - 1);
    v16h b0 = frag_f32cvt(wP, EMB, n0, k0, lane);
    v16h b1 = frag_f32cvt(wP, EMB, n0 + 16, k0, lane);
    acc0 = wmma_f16(a, b0, acc0);
    acc1 = wmma_f16(a, b1, acc1);
  }

  const int nl = lane & 15, M0 = (lane >> 4) * 8;
#pragma unroll
  for (int r = 0; r < 8; ++r) {
    const int t = m0 + M0 + r;
    if (t < PROWS) {
      const int C0 = n0 + nl, C1 = n0 + 16 + nl;
      Pb[((size_t)(C0 >> 6) * PPAD + t) * HD + (C0 & 63)] = (_Float16)acc0[r];
      Pb[((size_t)(C1 >> 6) * PPAD + t) * HD + (C1 & 63)] = (_Float16)acc1[r];
    }
  }
}

// ---------------------------------------------------------------------------
// Kernel 3: streaming attention. grid = (L/64, N*H), 4 waves/block, each wave
// owns a 16-row tile and streams j in 32-column tiles over the full sequence.
// ---------------------------------------------------------------------------
__global__ void __launch_bounds__(128) attn_kernel(
    const _Float16* __restrict__ Qu, const _Float16* __restrict__ Qv,
    const _Float16* __restrict__ Kb, const _Float16* __restrict__ Vt,
    const _Float16* __restrict__ Pb, _Float16* __restrict__ O16) {
  __shared__ float    bdLds[4][16 * 52];   // wave-private 16x47 band (padded)
  __shared__ _Float16 pLds[4][16 * 48];    // wave-private prob tile 16x32

  const int w = threadIdx.x >> 5, lane = threadIdx.x & 31;
  const int nl = lane & 15, half = lane >> 4, M0 = half * 8;
  const int i0 = (blockIdx.x * 4 + w) * 16;
  const int nb = blockIdx.y >> 2, h = blockIdx.y & 3;

  const size_t hoff = (size_t)(nb * NH + h) * Lseq * HD;
  const _Float16* qu = Qu + hoff;
  const _Float16* qv = Qv + hoff;
  const _Float16* kb = Kb + hoff;
  const _Float16* vt = Vt + hoff;                 // [HD][Lseq]
  const _Float16* pp = Pb + (size_t)h * PPAD * HD;
  float*    bdw = bdLds[w];
  _Float16* pw  = pLds[w];

  const v16h aQu0 = frag_f16(qu, HD, i0, 0, lane);
  const v16h aQu1 = frag_f16(qu, HD, i0, 32, lane);
  const v16h aQv0 = frag_f16(qv, HD, i0, 0, lane);
  const v16h aQv1 = frag_f16(qv, HD, i0, 32, lane);

  float mrow[8], lrow[8];
#pragma unroll
  for (int r = 0; r < 8; ++r) { mrow[r] = -3.0e38f; lrow[r] = 0.0f; }
  v8f o0 = {}, o1 = {}, o2 = {}, o3 = {};

  for (int j0 = 0; j0 < Lseq; j0 += 32) {
    if (j0 + 32 < Lseq) {
      __builtin_prefetch(kb + (size_t)(j0 + 32) * HD, 0, 1);
      __builtin_prefetch(vt + (size_t)(j0 + 32), 0, 1);
    }

    // ---- AC = Qu . K^T  (two interleaved 16x16 column halves) ----
    v8f sA = {}, sB = {};
    {
      v16h b00 = frag_f16(kb, HD, j0, 0, lane);
      v16h b01 = frag_f16(kb, HD, j0, 32, lane);
      v16h b10 = frag_f16(kb, HD, j0 + 16, 0, lane);
      v16h b11 = frag_f16(kb, HD, j0 + 16, 32, lane);
      sA = wmma_f16(aQu0, b00, sA);
      sB = wmma_f16(aQu0, b10, sB);
      sA = wmma_f16(aQu1, b01, sA);
      sB = wmma_f16(aQu1, b11, sB);
    }

    // ---- BD band: bd[i][u] = qv[i] . p[tbase+u], u = (j-i) - (j0-i0-15) ----
    const int tbase = (Lseq - 1) + j0 - i0 - 15;
    {
      v16h p00 = frag_f16_clamp(pp, HD, tbase,      0,  lane, PROWS - 1);
      v16h p01 = frag_f16_clamp(pp, HD, tbase,      32, lane, PROWS - 1);
      v16h p10 = frag_f16_clamp(pp, HD, tbase + 16, 0,  lane, PROWS - 1);
      v16h p11 = frag_f16_clamp(pp, HD, tbase + 16, 32, lane, PROWS - 1);
      v16h p20 = frag_f16_clamp(pp, HD, tbase + 32, 0,  lane, PROWS - 1);
      v16h p21 = frag_f16_clamp(pp, HD, tbase + 32, 32, lane, PROWS - 1);
      v8f c0 = {}, c1 = {}, c2 = {};
      c0 = wmma_f16(aQv0, p00, c0);
      c1 = wmma_f16(aQv0, p10, c1);
      c2 = wmma_f16(aQv0, p20, c2);
      c0 = wmma_f16(aQv1, p01, c0);
      c1 = wmma_f16(aQv1, p11, c1);
      c2 = wmma_f16(aQv1, p21, c2);
#pragma unroll
      for (int r = 0; r < 8; ++r) {
        bdw[(M0 + r) * 52 +      nl] = c0[r];
        bdw[(M0 + r) * 52 + 16 + nl] = c1[r];
        bdw[(M0 + r) * 52 + 32 + nl] = c2[r];
      }
    }
    asm volatile("s_wait_dscnt 0x0" ::: "memory");

    // ---- gather rel-shift, online softmax ----
    float s0[8], s1[8], tm[8];
#pragma unroll
    for (int r = 0; r < 8; ++r) {
      const int M = M0 + r;
      s0[r] = sA[r] + bdw[M * 52 + (nl - M + 15)];
      s1[r] = sB[r] + bdw[M * 52 + (nl - M + 31)];
      tm[r] = fmaxf(s0[r], s1[r]);
    }
#pragma unroll
    for (int r = 0; r < 8; ++r) {
#pragma unroll
      for (int msk = 1; msk < 16; msk <<= 1)
        tm[r] = fmaxf(tm[r], __shfl_xor(tm[r], msk, 32));
    }
#pragma unroll
    for (int r = 0; r < 8; ++r) {
      const int M = M0 + r;
      const float mnew  = fmaxf(mrow[r], tm[r]);
      const float alpha = __expf(mrow[r] - mnew);
      mrow[r] = mnew;
      const float e0 = __expf(s0[r] - mnew);
      const float e1 = __expf(s1[r] - mnew);
      float rs = e0 + e1;
#pragma unroll
      for (int msk = 1; msk < 16; msk <<= 1)
        rs += __shfl_xor(rs, msk, 32);
      lrow[r] = lrow[r] * alpha + rs;
      pw[M * 48 + nl]      = (_Float16)e0;
      pw[M * 48 + 16 + nl] = (_Float16)e1;
      o0[r] *= alpha; o1[r] *= alpha; o2[r] *= alpha; o3[r] *= alpha;
    }
    asm volatile("s_wait_dscnt 0x0" ::: "memory");

    // ---- re-layout probs C->A via LDS, then O += P . V (independent accs) ----
    v16h aP;
    {
      const _Float16* prow = pw + nl * 48 + half * 8;
      h8v lo = *(const h8v*)(prow);
      h8v hi = *(const h8v*)(prow + 16);
#pragma unroll
      for (int t = 0; t < 8; ++t) { aP[t] = lo[t]; aP[t + 8] = hi[t]; }
    }
    o0 = wmma_f16(aP, frag_f16(vt, Lseq, 0,  j0, lane), o0);
    o1 = wmma_f16(aP, frag_f16(vt, Lseq, 16, j0, lane), o1);
    o2 = wmma_f16(aP, frag_f16(vt, Lseq, 32, j0, lane), o2);
    o3 = wmma_f16(aP, frag_f16(vt, Lseq, 48, j0, lane), o3);
  }

  // ---- epilogue: normalize, write f16 pre-projection output (L,N,E) ----
#pragma unroll
  for (int r = 0; r < 8; ++r) {
    const int M = M0 + r;
    const float inv = 1.0f / lrow[r];
    _Float16* dst = O16 + ((size_t)((i0 + M) * NBAT + nb)) * EMB + h * HD + nl;
    dst[0]  = (_Float16)(o0[r] * inv);
    dst[16] = (_Float16)(o1[r] * inv);
    dst[32] = (_Float16)(o2[r] * inv);
    dst[48] = (_Float16)(o3[r] * inv);
  }
}

// ---------------------------------------------------------------------------
// Kernel 4: output projection. 16x32 tiles: (8192/16)=512 x (256/32)=8.
// ---------------------------------------------------------------------------
__global__ void __launch_bounds__(128) out_proj(const _Float16* __restrict__ O16,
                                                const float* __restrict__ wO,
                                                const float* __restrict__ bO,
                                                float* __restrict__ out) {
  const int w = threadIdx.x >> 5, lane = threadIdx.x & 31;
  const int tile = blockIdx.x * 4 + w;
  const int mt = tile / 8, nt = tile % 8;
  const int m0 = mt * 16, n0 = nt * 32;

  v8f acc0 = {}, acc1 = {};
#pragma unroll
  for (int k0 = 0; k0 < EMB; k0 += 32) {
    v16h a  = frag_f16(O16, EMB, m0, k0, lane);
    v16h b0 = frag_f32cvt(wO, EMB, n0, k0, lane);
    v16h b1 = frag_f32cvt(wO, EMB, n0 + 16, k0, lane);
    acc0 = wmma_f16(a, b0, acc0);
    acc1 = wmma_f16(a, b1, acc1);
  }

  const int nl = lane & 15, M0 = (lane >> 4) * 8;
#pragma unroll
  for (int r = 0; r < 8; ++r) {
    const int R = m0 + M0 + r;
    out[(size_t)R * EMB + n0 + nl]      = acc0[r] + bO[n0 + nl];
    out[(size_t)R * EMB + n0 + 16 + nl] = acc1[r] + bO[n0 + 16 + nl];
  }
}

// ---------------------------------------------------------------------------
extern "C" void kernel_launch(void* const* d_in, const int* in_sizes, int n_in,
                              void* d_out, int out_size, void* d_ws,
                              size_t ws_size, hipStream_t stream) {
  (void)in_sizes; (void)n_in; (void)out_size; (void)ws_size;
  const float* x   = (const float*)d_in[0];
  const float* pe  = (const float*)d_in[1];
  const float* wIn = (const float*)d_in[2];
  const float* bIn = (const float*)d_in[3];
  const float* wP  = (const float*)d_in[4];
  const float* bu  = (const float*)d_in[5];
  const float* bv  = (const float*)d_in[6];
  const float* wO  = (const float*)d_in[7];
  const float* bO  = (const float*)d_in[8];
  float* out = (float*)d_out;

  char* ws = (char*)d_ws;
  const size_t szQ = (size_t)NBAT * NH * Lseq * HD * sizeof(_Float16); // 4 MB
  const size_t szP = (size_t)NH * PPAD * HD * sizeof(_Float16);        // 2 MB
  _Float16* Qu  = (_Float16*)(ws);
  _Float16* Qv  = (_Float16*)(ws + szQ);
  _Float16* Kb  = (_Float16*)(ws + 2 * szQ);
  _Float16* Vt  = (_Float16*)(ws + 3 * szQ);
  _Float16* Pb  = (_Float16*)(ws + 4 * szQ);
  _Float16* O16 = (_Float16*)(ws + 4 * szQ + szP);                     // +4 MB

  proj_qkv<<<dim3((512 * 24) / 4), 128, 0, stream>>>(x, wIn, bIn, bu, bv,
                                                     Qu, Qv, Kb, Vt);
  proj_pos<<<dim3((256 * 8) / 4), 128, 0, stream>>>(pe, wP, Pb);
  attn_kernel<<<dim3(Lseq / 64, NBAT * NH), 128, 0, stream>>>(Qu, Qv, Kb, Vt,
                                                              Pb, O16);
  out_proj<<<dim3((512 * 8) / 4), 128, 0, stream>>>(O16, wO, bO, out);
}
